// GraphPartitionModule_36636071035261
// MI455X (gfx1250) — matrix operand order repaired and checked by reading, not compiled, not run on hardware
//
#include <hip/hip_runtime.h>
#include <hip/hip_bf16.h>
#include <cstdint>
#include <cstddef>

#define NN 100000
#define NE 1600000
#define INF_ 128
#define HF 64

typedef _Float16 f16x8  __attribute__((ext_vector_type(8)));
typedef _Float16 f16x16 __attribute__((ext_vector_type(16)));
typedef float    f32x8  __attribute__((ext_vector_type(8)));

// ---------------- small utility kernels ----------------
__global__ void zero_f32_kernel(float* __restrict__ p, int n) {
    int i = blockIdx.x * blockDim.x + threadIdx.x;
    if (i < n) p[i] = 0.0f;
}
__global__ void zero_u32_kernel(unsigned* __restrict__ p, int n) {
    int i = blockIdx.x * blockDim.x + threadIdx.x;
    if (i < n) p[i] = 0u;
}
__global__ void cvt_f16_kernel(const float* __restrict__ in, _Float16* __restrict__ out, int n) {
    int i = blockIdx.x * blockDim.x + threadIdx.x;
    if (i < n) out[i] = (_Float16)in[i];
}

// ---------------- quantile: count + radix select ----------------
__global__ __launch_bounds__(256)
void count_kernel(const float* __restrict__ pred, int ne, unsigned* __restrict__ counts) {
    __shared__ unsigned s[2];
    if (threadIdx.x < 2) s[threadIdx.x] = 0u;
    __syncthreads();
    for (int i = blockIdx.x * blockDim.x + threadIdx.x; i < ne; i += gridDim.x * blockDim.x) {
        if (pred[i] <= 0.0f) atomicAdd(&s[0], 1u);
        else                 atomicAdd(&s[1], 1u);
    }
    __syncthreads();
    if (threadIdx.x < 2 && s[threadIdx.x]) atomicAdd(&counts[threadIdx.x], s[threadIdx.x]);
}

__global__ void prep_ranks_kernel(const unsigned* __restrict__ counts,
                                  int* __restrict__ job_k, unsigned* __restrict__ job_prefix,
                                  float* __restrict__ fracs) {
    int nn = (int)counts[0];   // subset pred <= 0  (NEG, q = 0.2)
    int np = (int)counts[1];   // subset pred  > 0  (POS, q = 0.8)
    double pn = 0.2 * (double)(nn - 1);
    double pp = 0.8 * (double)(np - 1);
    int kn = (int)pn;  // floor, pn >= 0
    int kp = (int)pp;
    fracs[0] = (float)(pn - (double)kn);
    fracs[1] = (float)(pp - (double)kp);
    job_k[0] = kn;
    job_k[1] = (kn + 1 < nn) ? kn + 1 : nn - 1;
    job_k[2] = kp;
    job_k[3] = (kp + 1 < np) ? kp + 1 : np - 1;
    for (int j = 0; j < 4; ++j) job_prefix[j] = 0u;
}

__device__ __forceinline__ unsigned key_of(float v) {
    unsigned u = __float_as_uint(v);
    return (u & 0x80000000u) ? ~u : (u | 0x80000000u);
}
__device__ __forceinline__ float float_of_key(unsigned k) {
    unsigned u = (k & 0x80000000u) ? (k ^ 0x80000000u) : ~k;
    return __uint_as_float(u);
}

__global__ __launch_bounds__(256)
void radix_hist_kernel(const float* __restrict__ pred, int ne, int job, int pass,
                       const unsigned* __restrict__ job_prefix, unsigned* __restrict__ hist) {
    __shared__ unsigned sh[256];
    sh[threadIdx.x] = 0u;
    __syncthreads();
    const unsigned prefix = job_prefix[job];
    const int shift = 24 - pass * 8;
    const unsigned pm = pass ? (0xFFFFFFFFu << (shift + 8)) : 0u;
    const bool neg_subset = (job < 2);
    for (int i = blockIdx.x * blockDim.x + threadIdx.x; i < ne; i += gridDim.x * blockDim.x) {
        float v = pred[i];
        bool in = neg_subset ? (v <= 0.0f) : (v > 0.0f);
        if (!in) continue;
        unsigned key = key_of(v);
        if ((key & pm) != prefix) continue;
        atomicAdd(&sh[(key >> shift) & 255u], 1u);
    }
    __syncthreads();
    if (sh[threadIdx.x]) atomicAdd(&hist[threadIdx.x], sh[threadIdx.x]);
}

__global__ void radix_select_kernel(int job, int pass, unsigned* __restrict__ hist,
                                    int* __restrict__ job_k, unsigned* __restrict__ job_prefix,
                                    unsigned* __restrict__ sel_key) {
    __shared__ unsigned cnt[256];
    int t = threadIdx.x;
    cnt[t] = hist[t];
    hist[t] = 0u;      // reset for next pass
    __syncthreads();
    if (t == 0) {
        int k = job_k[job];
        unsigned cum = 0;
        int b = 0;
        for (; b < 256; ++b) {
            if (cum + cnt[b] > (unsigned)k) break;
            cum += cnt[b];
        }
        if (b == 256) b = 255;
        job_k[job] = k - (int)cum;
        int shift = 24 - pass * 8;
        unsigned pref = job_prefix[job] | ((unsigned)b << shift);
        job_prefix[job] = pref;
        if (pass == 3) sel_key[job] = pref;
    }
}

__global__ void finalize_thr_kernel(const unsigned* __restrict__ sel_key,
                                    const float* __restrict__ fracs, float* __restrict__ thr) {
    for (int s = 0; s < 2; ++s) {
        float lo = float_of_key(sel_key[2 * s]);
        float hi = float_of_key(sel_key[2 * s + 1]);
        thr[s] = lo + fracs[s] * (hi - lo);   // thr[0]=neg_thr, thr[1]=pos_thr
    }
}

// ---------------- WMMA GEMM:  out = relu(A[N,K] @ Bpacked[K,64] + bias) ----------------
// Bpacked layout: [kt][ntile(4)][lane(32)][16 halves], i<8: K=kt*32+8*(lane>>4)+i, i>=8: K=kt*32+16+8*(lane>>4)+(i-8), col=nt*16+(lane&15)
__global__ void pack_b_kernel(const float* __restrict__ W, _Float16* __restrict__ Bp, int K) {
    int i = blockIdx.x * blockDim.x + threadIdx.x;
    if (i >= K * 64) return;
    int idx = i;
    int h    = idx & 15;  idx >>= 4;
    int lane = idx & 31;  idx >>= 5;
    int nt   = idx & 3;   idx >>= 2;
    int kt   = idx;
    int krow = kt * 32 + ((lane >> 4) << 3) + (h & 7) + ((h >> 3) << 4);
    int col  = nt * 16 + (lane & 15);
    Bp[i] = (_Float16)W[krow * 64 + col];
}

__global__ __launch_bounds__(128)
void wmma_gemm_kernel(const _Float16* __restrict__ A, const _Float16* __restrict__ Bp,
                      const float* __restrict__ bias, float* __restrict__ out32,
                      _Float16* __restrict__ out16, int K) {
    const int lane = threadIdx.x & 31;
    const int wave = threadIdx.x >> 5;           // N tile 0..3
    const int rowBase = blockIdx.x * 16;
    const int row = rowBase + (lane & 15);
    const int halfSel = lane >> 4;               // 0 / 1
    const int nkt = K >> 5;
    f32x8 c = {};
    for (int kt = 0; kt < nkt; ++kt) {
        const int k0 = kt * 32 + halfSel * 8;
        f16x8 alo = *(const f16x8*)(A + (size_t)row * K + k0);
        f16x8 ahi = *(const f16x8*)(A + (size_t)row * K + k0 + 16);
        f16x16 a;
#pragma unroll
        for (int i = 0; i < 8; ++i) { a[i] = alo[i]; a[8 + i] = ahi[i]; }
        f16x16 b = *(const f16x16*)(Bp + ((size_t)(kt * 4 + wave) * 32 + lane) * 16);
        c = __builtin_amdgcn_wmma_f32_16x16x32_f16(false, a, false, b, (short)0, c, false, false);
    }
    const int col = wave * 16 + (lane & 15);
    const float bv = bias[col];
#pragma unroll
    for (int r = 0; r < 8; ++r) {
        int m = rowBase + halfSel * 8 + r;
        float v = fmaxf(c[r] + bv, 0.0f);
        if (out32) out32[(size_t)m * 64 + col] = v;
        if (out16) out16[(size_t)m * 64 + col] = (_Float16)v;
    }
}

// ---------------- graph propagation ----------------
__device__ __forceinline__ bool edge_in(float p, int mode, float negThr, float posThr) {
    if (mode == 0) return p < negThr;                     // m_neg
    if (mode == 1) return p > posThr;                     // m_pos
    return (p <= posThr) && (p >= negThr);                // m_unk
}

__global__ __launch_bounds__(256)
void deg_kernel(const float* __restrict__ pred, const int* __restrict__ dst, int ne,
                int mode, const float* __restrict__ thr, float* __restrict__ degv) {
    int e = blockIdx.x * blockDim.x + threadIdx.x;
    if (e >= ne) return;
    if (edge_in(pred[e], mode, thr[0], thr[1])) atomicAdd(&degv[dst[e]], 1.0f);
}

__global__ void dinv_kernel(float* __restrict__ degv, int n) {
    int i = blockIdx.x * blockDim.x + threadIdx.x;
    if (i < n) degv[i] = rsqrtf(fmaxf(degv[i], 1.0f));   // clip(deg,1)^-0.5, in-place
}

__global__ __launch_bounds__(256)
void edge_agg_kernel(const float* __restrict__ pred, const int* __restrict__ src,
                     const int* __restrict__ dst, int ne, int mode, const float* __restrict__ thr,
                     const float* __restrict__ f, const float* __restrict__ dinv,
                     float* __restrict__ agg) {
    int fi = threadIdx.x & 63;
    int e = blockIdx.x * 4 + (threadIdx.x >> 6);
    if (e >= ne) return;
    float p = pred[e];
    if (!edge_in(p, mode, thr[0], thr[1])) return;
    int s = src[e], d = dst[e];
    float v = f[(size_t)s * 64 + fi] * dinv[s];
    atomicAdd(&agg[(size_t)d * 64 + fi], v);
}

__global__ void lap_update_kernel(const float* __restrict__ fprev, const float* __restrict__ agg,
                                  const float* __restrict__ dinv, float* __restrict__ fnext, int n) {
    int i = blockIdx.x * blockDim.x + threadIdx.x;
    if (i < n) fnext[i] = fprev[i] - agg[i] * dinv[i >> 6];
}

__global__ void combo_kernel(const float* __restrict__ h, const float* __restrict__ L1,
                             const float* __restrict__ L2, float t0, float t1, float t2,
                             _Float16* __restrict__ hfinal, int colBase, int n) {
    int i = blockIdx.x * blockDim.x + threadIdx.x;
    if (i >= n) return;
    int node = i >> 6, fi = i & 63;
    float v = t0 * h[i] + t1 * L1[i] + t2 * L2[i];
    hfinal[(size_t)node * 192 + colBase + fi] = (_Float16)v;
}

// ---------------- driver ----------------
extern "C" void kernel_launch(void* const* d_in, const int* in_sizes, int n_in,
                              void* d_out, int out_size, void* d_ws, size_t ws_size,
                              hipStream_t stream) {
    const float* feat = (const float*)d_in[0];
    const float* pred = (const float*)d_in[1];
    const int*   src  = (const int*)d_in[2];
    const int*   dst  = (const int*)d_in[3];
    const float* W1 = (const float*)d_in[4];
    const float* b1 = (const float*)d_in[5];
    const float* W2 = (const float*)d_in[6];
    const float* b2 = (const float*)d_in[7];
    const float* W3 = (const float*)d_in[8];
    const float* b3 = (const float*)d_in[9];
    float* out = (float*)d_out;
    (void)in_sizes; (void)n_in; (void)out_size; (void)ws_size;

    char* ws = (char*)d_ws;
    size_t off = 0;
    auto take = [&](size_t bytes) -> char* {
        char* p = ws + off;
        off += (bytes + 255) & ~(size_t)255;
        return p;
    };
    _Float16* feat16 = (_Float16*)take((size_t)NN * INF_ * 2);   // 25.6 MB
    _Float16* h1_16  = (_Float16*)take((size_t)NN * HF * 2);     // 12.8 MB
    _Float16* hfinal16 = feat16;  // aliases feat16+h1_16 (exactly NN*192*2 bytes); both dead by then
    float* h32  = (float*)take((size_t)NN * HF * 4);
    float* L1   = (float*)take((size_t)NN * HF * 4);
    float* agg  = (float*)take((size_t)NN * HF * 4);
    float* degv = (float*)take((size_t)NN * 4);
    _Float16* pB1 = (_Float16*)take((size_t)INF_ * HF * 2);
    _Float16* pB2 = (_Float16*)take((size_t)HF * HF * 2);
    _Float16* pB3 = (_Float16*)take((size_t)3 * HF * HF * 2);
    unsigned* state = (unsigned*)take(2048);
    unsigned* counts     = state;               // 2
    int*      job_k      = (int*)(state + 2);   // 4
    unsigned* job_prefix = state + 6;           // 4
    float*    fracs      = (float*)(state + 10);// 2
    unsigned* sel_key    = state + 12;          // 4
    float*    thr        = (float*)(state + 16);// 2
    unsigned* hist       = state + 32;          // 256

    const int NH = NN * HF;   // 6.4M

    // 1) quantile thresholds (exact, linear interp like nanquantile)
    zero_u32_kernel<<<2, 256, 0, stream>>>(state, 288);
    count_kernel<<<512, 256, 0, stream>>>(pred, NE, counts);
    prep_ranks_kernel<<<1, 1, 0, stream>>>(counts, job_k, job_prefix, fracs);
    for (int job = 0; job < 4; ++job)
        for (int pass = 0; pass < 4; ++pass) {
            radix_hist_kernel<<<512, 256, 0, stream>>>(pred, NE, job, pass, job_prefix, hist);
            radix_select_kernel<<<1, 256, 0, stream>>>(job, pass, hist, job_k, job_prefix, sel_key);
        }
    finalize_thr_kernel<<<1, 1, 0, stream>>>(sel_key, fracs, thr);

    // 2) MLP via WMMA
    cvt_f16_kernel<<<(NN * INF_ + 255) / 256, 256, 0, stream>>>(feat, feat16, NN * INF_);
    pack_b_kernel<<<(INF_ * 64 + 255) / 256, 256, 0, stream>>>(W1, pB1, INF_);
    pack_b_kernel<<<(HF * 64 + 255) / 256, 256, 0, stream>>>(W2, pB2, HF);
    pack_b_kernel<<<(3 * HF * 64 + 255) / 256, 256, 0, stream>>>(W3, pB3, 3 * HF);
    wmma_gemm_kernel<<<NN / 16, 128, 0, stream>>>(feat16, pB1, b1, nullptr, h1_16, INF_);
    wmma_gemm_kernel<<<NN / 16, 128, 0, stream>>>(h1_16, pB2, b2, h32, nullptr, HF);

    // 3) Laplacian powers + theta combos; T rows: pos=T[2], unk=T[1], neg=T[0]
    const int   modes[3]  = {1, 2, 0};          // pos, unk, neg
    const int   colb[3]   = {0, 64, 128};
    const float t0s[3]    = {0.0f, 0.0f, 3.0f};
    const float t1s[3]    = {0.0f, 3.0f, -3.0f};
    const float t2s[3]    = {0.75f, -1.5f, 0.75f};
    for (int mi = 0; mi < 3; ++mi) {
        int mode = modes[mi];
        zero_f32_kernel<<<(NN + 255) / 256, 256, 0, stream>>>(degv, NN);
        deg_kernel<<<(NE + 255) / 256, 256, 0, stream>>>(pred, dst, NE, mode, thr, degv);
        dinv_kernel<<<(NN + 255) / 256, 256, 0, stream>>>(degv, NN);
        // L1 = h - D^-1/2 A D^-1/2 h
        zero_f32_kernel<<<(NH + 255) / 256, 256, 0, stream>>>(agg, NH);
        edge_agg_kernel<<<NE / 4, 256, 0, stream>>>(pred, src, dst, NE, mode, thr, h32, degv, agg);
        lap_update_kernel<<<(NH + 255) / 256, 256, 0, stream>>>(h32, agg, degv, L1, NH);
        // L2 = L1 - D^-1/2 A D^-1/2 L1   (written into agg in place)
        zero_f32_kernel<<<(NH + 255) / 256, 256, 0, stream>>>(agg, NH);
        edge_agg_kernel<<<NE / 4, 256, 0, stream>>>(pred, src, dst, NE, mode, thr, L1, degv, agg);
        lap_update_kernel<<<(NH + 255) / 256, 256, 0, stream>>>(L1, agg, degv, agg, NH);
        combo_kernel<<<(NH + 255) / 256, 256, 0, stream>>>(h32, L1, agg, t0s[mi], t1s[mi], t2s[mi],
                                                           hfinal16, colb[mi], NH);
    }

    // 4) final GEMM -> relu -> d_out
    wmma_gemm_kernel<<<NN / 16, 128, 0, stream>>>(hfinal16, pB3, b3, out, nullptr, 3 * HF);
}
